// GraphSAGEModel_13108240187440
// MI455X (gfx1250) — compile-verified, matrix operand order checked by reading, not run on hardware
//
#include <hip/hip_runtime.h>

#define NNODES 100000
#define NEDGES 1600000
#define DIM    128
#define NGRAPH 64
#define EPSBN  1e-5f

typedef __attribute__((ext_vector_type(16))) __bf16 v16bf;
typedef __attribute__((ext_vector_type(8)))  float  v8f;

// ---------------- workspace layout (bytes, all offsets 256B aligned) ----------------
constexpr size_t SZ_H    = (size_t)NNODES * DIM * 4;      // 51,200,000
constexpr size_t OFF_HA  = 0;
constexpr size_t OFF_HB  = OFF_HA + SZ_H;
constexpr size_t OFF_AGG = OFF_HB + SZ_H;
constexpr size_t OFF_DEG = OFF_AGG + SZ_H;                // N*4 (padded)
constexpr size_t OFF_INV = OFF_DEG + 400384;
constexpr size_t OFF_WB  = OFF_INV + 400384;              // bf16 swizzled weights: 16*16384*2
constexpr size_t OFF_BNS = OFF_WB  + 524288;
constexpr size_t OFF_BNQ = OFF_BNS + 512;
constexpr size_t OFF_PS  = OFF_BNQ + 512;                 // 64*128*4
constexpr size_t OFF_PM  = OFF_PS  + 32768;
constexpr size_t OFF_PC  = OFF_PM  + 32768;               // 64*4
constexpr size_t OFF_G   = OFF_PC  + 256;                 // 64*256*4
constexpr size_t OFF_Z1  = OFF_G   + 65536;               // 64*128*4
constexpr size_t OFF_Z2  = OFF_Z1  + 32768;               // 64*64*4

// ---------------- helpers ----------------
__device__ __forceinline__ unsigned short f2bf(float f) {  // RNE float -> bf16 bits
  unsigned u = __float_as_uint(f);
  u += 0x7fffu + ((u >> 16) & 1u);
  return (unsigned short)(u >> 16);
}

union BFrag { uint4 q[2]; v16bf v; };

// Build the 16x32 bf16 A-fragment (scaled): native casts -> packed cvt instructions.
// lane (l&15) = row-in-tile, kbase = kt*32 + (l>>4)*8; halves at +0..7 and +16..23.
__device__ __forceinline__ v16bf load_a_frag_scaled(const float* __restrict__ rowp,
                                                    int kbase, float s) {
  const float4 f0 = *(const float4*)(rowp + kbase);
  const float4 f1 = *(const float4*)(rowp + kbase + 4);
  const float4 f2 = *(const float4*)(rowp + kbase + 16);
  const float4 f3 = *(const float4*)(rowp + kbase + 20);
  v16bf a;
  a[0]  = (__bf16)(f0.x * s); a[1]  = (__bf16)(f0.y * s);
  a[2]  = (__bf16)(f0.z * s); a[3]  = (__bf16)(f0.w * s);
  a[4]  = (__bf16)(f1.x * s); a[5]  = (__bf16)(f1.y * s);
  a[6]  = (__bf16)(f1.z * s); a[7]  = (__bf16)(f1.w * s);
  a[8]  = (__bf16)(f2.x * s); a[9]  = (__bf16)(f2.y * s);
  a[10] = (__bf16)(f2.z * s); a[11] = (__bf16)(f2.w * s);
  a[12] = (__bf16)(f3.x * s); a[13] = (__bf16)(f3.y * s);
  a[14] = (__bf16)(f3.z * s); a[15] = (__bf16)(f3.w * s);
  return a;
}

__device__ __forceinline__ v16bf load_a_frag(const float* __restrict__ rowp, int kbase) {
  const float4 f0 = *(const float4*)(rowp + kbase);
  const float4 f1 = *(const float4*)(rowp + kbase + 4);
  const float4 f2 = *(const float4*)(rowp + kbase + 16);
  const float4 f3 = *(const float4*)(rowp + kbase + 20);
  v16bf a;
  a[0]  = (__bf16)f0.x; a[1]  = (__bf16)f0.y;
  a[2]  = (__bf16)f0.z; a[3]  = (__bf16)f0.w;
  a[4]  = (__bf16)f1.x; a[5]  = (__bf16)f1.y;
  a[6]  = (__bf16)f1.z; a[7]  = (__bf16)f1.w;
  a[8]  = (__bf16)f2.x; a[9]  = (__bf16)f2.y;
  a[10] = (__bf16)f2.z; a[11] = (__bf16)f2.w;
  a[12] = (__bf16)f3.x; a[13] = (__bf16)f3.y;
  a[14] = (__bf16)f3.z; a[15] = (__bf16)f3.w;
  return a;
}

// ---------------- small utility kernels ----------------
__global__ void k_zero(float* __restrict__ p, int n) {
  int i = blockIdx.x * 256 + threadIdx.x;
  if (i < n) p[i] = 0.f;
}

__global__ void k_zero4(float4* __restrict__ p, int n4) {
  int i = blockIdx.x * 256 + threadIdx.x;
  if (i < n4) p[i] = make_float4(0.f, 0.f, 0.f, 0.f);
}

__global__ void k_deg(const int* __restrict__ ei, float* __restrict__ deg) {
  int e = blockIdx.x * 256 + threadIdx.x;
  if (e < NEDGES) atomicAdd(&deg[ei[NEDGES + e]], 1.f);
}

__global__ void k_inv(const float* __restrict__ deg, float* __restrict__ inv) {
  int i = blockIdx.x * 256 + threadIdx.x;
  if (i < NNODES) inv[i] = 1.f / fmaxf(deg[i], 1.f);
}

// Pre-swizzle W into the WMMA B-fragment layout (bf16). One element per thread.
// Per 128x128 matrix: 32 tiles (kt 0..3, nt 0..7); tile = 32 lanes x 16 bf16 contiguous.
__global__ void k_wconv(const float* __restrict__ Wl, const float* __restrict__ Wr,
                        unsigned short* __restrict__ wb) {
  int idx = blockIdx.x * 256 + threadIdx.x;          // 16 * 16384
  int m    = idx >> 14;                              // matrix id: layer*2 + (0=Wl,1=Wr)
  int rem  = idx & 16383;
  int tile = rem >> 9;                               // kt*8 + nt
  int lr   = rem & 511;
  int lane = lr >> 4;
  int j    = lr & 15;
  int kt = tile >> 3, nt = tile & 7;
  int n  = nt * 16 + (lane & 15);
  int kb = kt * 32 + (lane >> 4) * 8;
  int k  = kb + (j < 8 ? j : 8 + j);                 // +0..7 then +16..23
  int layer = m >> 1;
  const float* W = (m & 1) ? Wr : Wl;
  wb[idx] = f2bf(W[(size_t)layer * DIM * DIM + (size_t)k * DIM + n]);
}

// Edge scatter: one lane-quarter per (edge, 4-float chunk). Coalesced float4 gather,
// 4x global_atomic_add_f32 scatter.
__global__ void k_scatter(const float* __restrict__ h, const int* __restrict__ ei,
                          float* __restrict__ agg) {
  unsigned idx = blockIdx.x * 256u + threadIdx.x;    // E*32 work items
  if (idx >= (unsigned)NEDGES * 32u) return;
  int e  = idx >> 5;
  int c4 = idx & 31;
  int s = ei[e];
  int d = ei[NEDGES + e];
  float4 v = ((const float4*)(h + (size_t)s * DIM))[c4];
  float* a = agg + (size_t)d * DIM + c4 * 4;
  atomicAdd(a + 0, v.x);
  atomicAdd(a + 1, v.y);
  atomicAdd(a + 2, v.z);
  atomicAdd(a + 3, v.w);
}

// ---------------- fused SAGEConv GEMM: hout = (agg*inv_deg)@Wl + b + hin@Wr ----------------
// Block = 256 threads = 8 waves; block covers 128 rows x 128 cols.
// Weights staged in LDS (32KB per phase, 16B lane-granules to minimize bank conflicts);
// each wave: 16 rows, 8 column tiles, 2 x K=128 -> 64 v_wmma_f32_16x16x32_bf16.
__global__ __launch_bounds__(256) void k_sage_gemm(
    const float* __restrict__ hin, const float* __restrict__ agg,
    const float* __restrict__ invdeg,
    const unsigned short* __restrict__ wl_bf, const unsigned short* __restrict__ wr_bf,
    const float* __restrict__ bias, float* __restrict__ hout, int do_relu) {
  // [tile*2 + half][lane] : 16B granules, lane stride 16B
  __shared__ uint4 ldsW[64][32];

  const int tid  = threadIdx.x;
  const int lane = tid & 31;
  const int wave = tid >> 5;
  const int rowBase = blockIdx.x * 128 + wave * 16;
  const int mrow  = lane & 15;
  const int khalf = lane >> 4;

  int r  = rowBase + mrow;
  int rc = r < NNODES ? r : NNODES - 1;              // clamp loads; stores predicated
  const float scale = invdeg[rc];
  const size_t rowOff = (size_t)rc * DIM;

  v8f acc[8] = {};

  // ---- phase 1: (agg * inv_deg) @ Wl ----
  {
    const uint4* src = (const uint4*)wl_bf;          // global granule g = tile*64 + lane*2 + half
    for (int g = tid; g < 2048; g += 256)
      ldsW[((g >> 6) << 1) | (g & 1)][(g & 63) >> 1] = src[g];
  }
  __syncthreads();
#pragma unroll
  for (int kt = 0; kt < 4; ++kt) {
    v16bf a = load_a_frag_scaled(agg + rowOff, kt * 32 + khalf * 8, scale);
#pragma unroll
    for (int nt = 0; nt < 8; ++nt) {
      const int t = kt * 8 + nt;
      BFrag b;
      b.q[0] = ldsW[t * 2][lane];
      b.q[1] = ldsW[t * 2 + 1][lane];
      acc[nt] = __builtin_amdgcn_wmma_f32_16x16x32_bf16(false, a, false, b.v,
                                                        (short)0, acc[nt], false, false);
    }
  }
  __syncthreads();

  // ---- phase 2: hin @ Wr ----
  {
    const uint4* src = (const uint4*)wr_bf;
    for (int g = tid; g < 2048; g += 256)
      ldsW[((g >> 6) << 1) | (g & 1)][(g & 63) >> 1] = src[g];
  }
  __syncthreads();
#pragma unroll
  for (int kt = 0; kt < 4; ++kt) {
    v16bf a = load_a_frag(hin + rowOff, kt * 32 + khalf * 8);
#pragma unroll
    for (int nt = 0; nt < 8; ++nt) {
      const int t = kt * 8 + nt;
      BFrag b;
      b.q[0] = ldsW[t * 2][lane];
      b.q[1] = ldsW[t * 2 + 1][lane];
      acc[nt] = __builtin_amdgcn_wmma_f32_16x16x32_bf16(false, a, false, b.v,
                                                        (short)0, acc[nt], false, false);
    }
  }

  // epilogue: C layout -> VGPR rr holds M = rr + khalf*8, N = lane&15
#pragma unroll
  for (int nt = 0; nt < 8; ++nt) {
    int col = nt * 16 + (lane & 15);
    float bcol = bias[col];
#pragma unroll
    for (int rr = 0; rr < 8; ++rr) {
      int row = rowBase + khalf * 8 + rr;
      float v = acc[nt][rr] + bcol;
      if (do_relu) v = fmaxf(v, 0.f);
      if (row < NNODES) hout[(size_t)row * DIM + col] = v;
    }
  }
}

// ---------------- BatchNorm ----------------
__global__ void k_bn_stats(const float* __restrict__ h, float* __restrict__ s,
                           float* __restrict__ q) {
  int col = threadIdx.x & 127;
  int sub = threadIdx.x >> 7;
  float ls = 0.f, lq = 0.f;
  for (int row = blockIdx.x * 2 + sub; row < NNODES; row += gridDim.x * 2) {
    float v = h[(size_t)row * DIM + col];
    ls += v; lq += v * v;
  }
  atomicAdd(&s[col], ls);
  atomicAdd(&q[col], lq);
}

__global__ void k_bn_apply(float* __restrict__ h, const float* __restrict__ s,
                           const float* __restrict__ q, const float* __restrict__ gamma,
                           const float* __restrict__ beta) {
  int idx = blockIdx.x * 256 + threadIdx.x;
  if (idx >= NNODES * DIM) return;
  int col = idx & 127;
  const float rn = 1.f / (float)NNODES;
  float mu  = s[col] * rn;
  float var = q[col] * rn - mu * mu;
  float v = (h[idx] - mu) * rsqrtf(var + EPSBN) * gamma[col] + beta[col];
  h[idx] = fmaxf(v, 0.f);
}

// ---------------- pooling + classifier ----------------
// h is post-ReLU (>=0), so u32 atomicMax on float bits is order-preserving.
__global__ void k_pool(const float* __restrict__ h, const int* __restrict__ batch,
                       float* __restrict__ psum, float* __restrict__ pmax,
                       float* __restrict__ pcnt) {
  int idx = blockIdx.x * 256 + threadIdx.x;
  if (idx >= NNODES * DIM) return;
  int row = idx >> 7, col = idx & 127;
  int b = batch[row];
  float v = h[idx];
  atomicAdd(&psum[b * DIM + col], v);
  atomicMax((unsigned int*)&pmax[b * DIM + col], __float_as_uint(v));
  if (col == 0) atomicAdd(&pcnt[b], 1.f);
}

__global__ void k_gbuild(const float* __restrict__ psum, const float* __restrict__ pmax,
                         const float* __restrict__ pcnt, float* __restrict__ g) {
  int idx = blockIdx.x * 256 + threadIdx.x;          // 64*256
  int gi = idx >> 8, c = idx & 255;
  float v = (c < DIM) ? psum[gi * DIM + c] / fmaxf(pcnt[gi], 1.f)
                      : pmax[gi * DIM + c - DIM];
  g[idx] = v;
}

__global__ void k_mlp1(const float* __restrict__ g, const float* __restrict__ W1,
                       const float* __restrict__ b1, float* __restrict__ z1) {
  int idx = blockIdx.x * 256 + threadIdx.x;          // 64*128
  int gi = idx >> 7, c = idx & 127;
  float s = b1[c];
  const float* gr = g + gi * 256;
  for (int k = 0; k < 256; ++k) s += gr[k] * W1[k * 128 + c];
  z1[idx] = fmaxf(s, 0.f);
}

__global__ void k_mlp2(const float* __restrict__ z1, const float* __restrict__ W2,
                       const float* __restrict__ b2, float* __restrict__ z2) {
  int idx = blockIdx.x * 256 + threadIdx.x;          // 64*64
  int gi = idx >> 6, c = idx & 63;
  float s = b2[c];
  const float* zr = z1 + gi * 128;
  for (int k = 0; k < 128; ++k) s += zr[k] * W2[k * 64 + c];
  z2[idx] = fmaxf(s, 0.f);
}

__global__ void k_mlp3(const float* __restrict__ z2, const float* __restrict__ W3,
                       const float* __restrict__ b3, float* __restrict__ out) {
  int gi = threadIdx.x;
  if (gi >= NGRAPH) return;
  float s = b3[0];
  const float* zr = z2 + gi * 64;
  for (int k = 0; k < 64; ++k) s += zr[k] * W3[k];
  out[gi] = s;
}

// ---------------- host launcher ----------------
extern "C" void kernel_launch(void* const* d_in, const int* in_sizes, int n_in,
                              void* d_out, int out_size, void* d_ws, size_t ws_size,
                              hipStream_t stream) {
  (void)in_sizes; (void)n_in; (void)out_size; (void)ws_size;
  const float* x     = (const float*)d_in[0];
  const int*   ei    = (const int*)d_in[1];
  const int*   batch = (const int*)d_in[2];
  const float* Wl    = (const float*)d_in[3];
  const float* bl    = (const float*)d_in[4];
  const float* Wr    = (const float*)d_in[5];
  const float* gamma = (const float*)d_in[6];
  const float* beta  = (const float*)d_in[7];
  const float* W1    = (const float*)d_in[8];
  const float* b1    = (const float*)d_in[9];
  const float* W2    = (const float*)d_in[10];
  const float* b2    = (const float*)d_in[11];
  const float* W3    = (const float*)d_in[12];
  const float* b3    = (const float*)d_in[13];

  char* ws = (char*)d_ws;
  float*          hA   = (float*)(ws + OFF_HA);
  float*          hB   = (float*)(ws + OFF_HB);
  float*          agg  = (float*)(ws + OFF_AGG);
  float*          deg  = (float*)(ws + OFF_DEG);
  float*          inv  = (float*)(ws + OFF_INV);
  unsigned short* wb   = (unsigned short*)(ws + OFF_WB);
  float*          bns  = (float*)(ws + OFF_BNS);
  float*          bnq  = (float*)(ws + OFF_BNQ);
  float*          psum = (float*)(ws + OFF_PS);
  float*          pmax = (float*)(ws + OFF_PM);
  float*          pcnt = (float*)(ws + OFF_PC);
  float*          g    = (float*)(ws + OFF_G);
  float*          z1   = (float*)(ws + OFF_Z1);
  float*          z2   = (float*)(ws + OFF_Z2);

  const int ND  = NNODES * DIM;                 // 12,800,000
  const int ND4 = ND / 4;                       // 3,200,000

  // degree + inverse degree
  k_zero<<<(NNODES + 255) / 256, 256, 0, stream>>>(deg, NNODES);
  k_deg<<<(NEDGES + 255) / 256, 256, 0, stream>>>(ei, deg);
  k_inv<<<(NNODES + 255) / 256, 256, 0, stream>>>(deg, inv);

  // swizzle all 16 weight matrices to bf16 WMMA B-fragments
  k_wconv<<<(16 * 16384) / 256, 256, 0, stream>>>(Wl, Wr, wb);

  const float* hin = x;
  float* bufs[2] = {hA, hB};
  for (int l = 0; l < 8; ++l) {
    float* hout = bufs[l & 1];
    // mean-aggregate neighbors
    k_zero4<<<ND4 / 256, 256, 0, stream>>>((float4*)agg, ND4);
    k_scatter<<<(NEDGES * 32) / 256, 256, 0, stream>>>(hin, ei, agg);
    // fused dual-GEMM (WMMA) + bias (+ReLU on first conv of each block)
    k_sage_gemm<<<(NNODES + 127) / 128, 256, 0, stream>>>(
        hin, agg, inv, wb + (size_t)(2 * l) * 16384, wb + (size_t)(2 * l + 1) * 16384,
        bl + l * DIM, hout, (l & 1) ? 0 : 1);
    if (l & 1) {  // BatchNorm + ReLU after second conv of the block
      int blk = l >> 1;
      k_zero<<<1, 256, 0, stream>>>(bns, DIM);
      k_zero<<<1, 256, 0, stream>>>(bnq, DIM);
      k_bn_stats<<<512, 256, 0, stream>>>(hout, bns, bnq);
      k_bn_apply<<<ND / 256, 256, 0, stream>>>(hout, bns, bnq, gamma + blk * DIM,
                                               beta + blk * DIM);
    }
    hin = hout;
  }

  // global mean/max pooling
  k_zero<<<(NGRAPH * DIM) / 256, 256, 0, stream>>>(psum, NGRAPH * DIM);
  k_zero<<<(NGRAPH * DIM) / 256, 256, 0, stream>>>(pmax, NGRAPH * DIM);
  k_zero<<<1, 256, 0, stream>>>(pcnt, NGRAPH);
  k_pool<<<ND / 256, 256, 0, stream>>>(hin, batch, psum, pmax, pcnt);
  k_gbuild<<<(NGRAPH * 256) / 256, 256, 0, stream>>>(psum, pmax, pcnt, g);

  // classifier head
  k_mlp1<<<(NGRAPH * 128) / 256, 256, 0, stream>>>(g, W1, b1, z1);
  k_mlp2<<<(NGRAPH * 64) / 256, 256, 0, stream>>>(z1, W2, b2, z2);
  k_mlp3<<<1, NGRAPH, 0, stream>>>(z2, W3, b3, (float*)d_out);
}
